// BitnetMLP_82867099009123
// MI455X (gfx1250) — 8-bit matrix operand order repaired and checked by reading, compiled, not run on hardware
//
#include <hip/hip_runtime.h>
#include <hip/hip_bf16.h>
#include <stdint.h>

typedef int v8i __attribute__((ext_vector_type(8)));

#define H_DIM 4096
#define I_DIM 11008
#define T_DIM 4096
#define NTHREADS 256

__device__ __forceinline__ void async_load_b128(void* lds, const void* gptr) {
  uint32_t lds_off = (uint32_t)(uintptr_t)lds;
  asm volatile("global_load_async_to_lds_b128 %0, %1, off"
               :: "v"(lds_off), "v"(gptr)
               : "memory");
}
#define ASYNC_WAIT(imm) asm volatile("s_wait_asynccnt " #imm ::: "memory")

__global__ void absmean_partial_kernel(const float* __restrict__ w, long n,
                                       float* __restrict__ part) {
  __shared__ float red[NTHREADS];
  float s = 0.f;
  for (long i = (long)blockIdx.x * NTHREADS + threadIdx.x; i < n;
       i += (long)NTHREADS * gridDim.x)
    s += fabsf(w[i]);
  red[threadIdx.x] = s;
  __syncthreads();
  for (int off = NTHREADS / 2; off > 0; off >>= 1) {
    if ((int)threadIdx.x < off) red[threadIdx.x] += red[threadIdx.x + off];
    __syncthreads();
  }
  if (threadIdx.x == 0) part[blockIdx.x] = red[0];
}

__global__ void absmean_final_kernel(const float* __restrict__ part, long n,
                                     float* __restrict__ wsc) {
  __shared__ float red[NTHREADS];
  const float* p = part + (long)blockIdx.x * 1024;
  float s = 0.f;
  for (int i = threadIdx.x; i < 1024; i += NTHREADS) s += p[i];
  red[threadIdx.x] = s;
  __syncthreads();
  for (int off = NTHREADS / 2; off > 0; off >>= 1) {
    if ((int)threadIdx.x < off) red[threadIdx.x] += red[threadIdx.x + off];
    __syncthreads();
  }
  if (threadIdx.x == 0) {
    float mean = red[0] / (float)n;
    wsc[blockIdx.x] = fmaxf(mean, 1e-5f);
  }
}

__global__ void quant_w_kernel(const float* __restrict__ w,
                               int8_t* __restrict__ wq, long n,
                               const float* __restrict__ dq) {
  const float inv = 1.0f / dq[0];
  for (long i = (long)blockIdx.x * NTHREADS + threadIdx.x; i < n;
       i += (long)NTHREADS * gridDim.x) {
    float q = rintf(w[i] * inv);
    q = fminf(fmaxf(q, -1.f), 1.f);
    wq[i] = (int8_t)(int)q;
  }
}

__global__ void actquant_x_kernel(const float* __restrict__ x,
                                  int8_t* __restrict__ xq,
                                  float* __restrict__ sx) {
  __shared__ float red[NTHREADS];
  __shared__ float s_scale;
  const int row = blockIdx.x;
  const float* xr = x + (long)row * H_DIM;
  float m = 0.f;
  for (int i = threadIdx.x; i < H_DIM; i += NTHREADS) m = fmaxf(m, fabsf(xr[i]));
  red[threadIdx.x] = m;
  __syncthreads();
  for (int off = NTHREADS / 2; off > 0; off >>= 1) {
    if ((int)threadIdx.x < off)
      red[threadIdx.x] = fmaxf(red[threadIdx.x], red[threadIdx.x + off]);
    __syncthreads();
  }
  if (threadIdx.x == 0) {
    float amax = fmaxf(red[0], 1e-5f);
    s_scale = 127.f / amax;
    sx[row] = amax / 127.f;
  }
  __syncthreads();
  const float sc = s_scale;
  int8_t* xqr = xq + (long)row * H_DIM;
  for (int i = threadIdx.x; i < H_DIM; i += NTHREADS) {
    float q = rintf(xr[i] * sc);
    q = fminf(fmaxf(q, -128.f), 127.f);
    xqr[i] = (int8_t)(int)q;
  }
}

__device__ __forceinline__ v8i load_a_frag(const int8_t* __restrict__ tile,
                                           int row0, int lane) {
  const int8_t* p = tile + (row0 + (lane & 15)) * 64 + ((lane >> 4) << 4);
  uint64_t q0 = *(const uint64_t*)(p + 0);
  uint64_t q1 = *(const uint64_t*)(p + 8);
  uint64_t q2 = *(const uint64_t*)(p + 32);
  uint64_t q3 = *(const uint64_t*)(p + 40);
  v8i f;
  f[0] = (int)(uint32_t)q0; f[1] = (int)(uint32_t)(q0 >> 32);
  f[2] = (int)(uint32_t)q1; f[3] = (int)(uint32_t)(q1 >> 32);
  f[4] = (int)(uint32_t)q2; f[5] = (int)(uint32_t)(q2 >> 32);
  f[6] = (int)(uint32_t)q3; f[7] = (int)(uint32_t)(q3 >> 32);
  return f;
}

__device__ __forceinline__ v8i load_b_frag(const int8_t* __restrict__ tile,
                                           int col0, int lane) {
  const int8_t* p = tile + (col0 + (lane & 15)) * 64 + ((lane >> 4) << 4);
  uint4 q0 = *(const uint4*)(p + 0);
  uint4 q1 = *(const uint4*)(p + 32);
  v8i f;
  f[0] = (int)q0.x; f[1] = (int)q0.y; f[2] = (int)q0.z; f[3] = (int)q0.w;
  f[4] = (int)q1.x; f[5] = (int)q1.y; f[6] = (int)q1.z; f[7] = (int)q1.w;
  return f;
}

__global__ __launch_bounds__(NTHREADS)
void gemm_gateup_kernel(const int8_t* __restrict__ xq,
                        const int8_t* __restrict__ wqg,
                        const int8_t* __restrict__ wqu,
                        const float* __restrict__ sx,
                        const float* __restrict__ wsc,
                        float* __restrict__ h) {
  __shared__ int8_t As[2][128 * 64];
  __shared__ int8_t Bg[2][64 * 64];
  __shared__ int8_t Bu[2][64 * 64];

  const int tid = threadIdx.x;
  const int lane = tid & 31;
  const int wave = tid >> 5;
  const int wm = wave >> 1;
  const int wn = wave & 1;
  const long m0 = (long)blockIdx.y * 128;
  const long n0 = (long)blockIdx.x * 64;

  const int ra0 = (tid) >> 2,            qa0 = (tid) & 3;
  const int ra1 = (tid + NTHREADS) >> 2, qa1 = (tid + NTHREADS) & 3;
  const int rb  = tid >> 2,              qb  = tid & 3;

  v8i accg[2][2] = {};
  v8i accu[2][2] = {};

  const int nsteps = H_DIM / 64;

  {
    const int k0 = 0;
    async_load_b128(&As[0][ra0 * 64 + qa0 * 16], &xq[(m0 + ra0) * (long)H_DIM + k0 + qa0 * 16]);
    async_load_b128(&As[0][ra1 * 64 + qa1 * 16], &xq[(m0 + ra1) * (long)H_DIM + k0 + qa1 * 16]);
    async_load_b128(&Bg[0][rb * 64 + qb * 16],  &wqg[(n0 + rb) * (long)H_DIM + k0 + qb * 16]);
    async_load_b128(&Bu[0][rb * 64 + qb * 16],  &wqu[(n0 + rb) * (long)H_DIM + k0 + qb * 16]);
  }

  for (int it = 0; it < nsteps; ++it) {
    const int cur = it & 1;
    if (it + 1 < nsteps) {
      const int nxt = cur ^ 1;
      const int k1 = (it + 1) * 64;
      async_load_b128(&As[nxt][ra0 * 64 + qa0 * 16], &xq[(m0 + ra0) * (long)H_DIM + k1 + qa0 * 16]);
      async_load_b128(&As[nxt][ra1 * 64 + qa1 * 16], &xq[(m0 + ra1) * (long)H_DIM + k1 + qa1 * 16]);
      async_load_b128(&Bg[nxt][rb * 64 + qb * 16],  &wqg[(n0 + rb) * (long)H_DIM + k1 + qb * 16]);
      async_load_b128(&Bu[nxt][rb * 64 + qb * 16],  &wqu[(n0 + rb) * (long)H_DIM + k1 + qb * 16]);
      ASYNC_WAIT(4);
    } else {
      ASYNC_WAIT(0);
    }
    __syncthreads();

    v8i a[2], bgf[2], buf[2];
    #pragma unroll
    for (int mt = 0; mt < 2; ++mt) a[mt] = load_a_frag(As[cur], wm * 32 + mt * 16, lane);
    #pragma unroll
    for (int nt = 0; nt < 2; ++nt) {
      bgf[nt] = load_b_frag(Bg[cur], wn * 32 + nt * 16, lane);
      buf[nt] = load_b_frag(Bu[cur], wn * 32 + nt * 16, lane);
    }
    #pragma unroll
    for (int mt = 0; mt < 2; ++mt)
      #pragma unroll
      for (int nt = 0; nt < 2; ++nt) {
        accg[mt][nt] = __builtin_amdgcn_wmma_i32_16x16x64_iu8(
            true, a[mt], true, bgf[nt], accg[mt][nt], false, false);
        accu[mt][nt] = __builtin_amdgcn_wmma_i32_16x16x64_iu8(
            true, a[mt], true, buf[nt], accu[mt][nt], false, false);
      }
    __syncthreads();
  }

  const float dqg = wsc[0];
  const float dqu = wsc[1];
  #pragma unroll
  for (int mt = 0; mt < 2; ++mt)
    #pragma unroll
    for (int nt = 0; nt < 2; ++nt) {
      const long rowbase = m0 + wm * 32 + mt * 16 + ((lane >> 4) << 3);
      const long col = n0 + wn * 32 + nt * 16 + (lane & 15);
      #pragma unroll
      for (int r = 0; r < 8; ++r) {
        long row = rowbase + r;
        float ax = sx[row];
        float g = (float)accg[mt][nt][r] * ax * dqg;
        float u = (float)accu[mt][nt][r] * ax * dqu;
        float sg = g / (1.f + __expf(-g));
        h[row * (long)I_DIM + col] = sg * u;
      }
    }
}

__global__ void rms_quant_kernel(const float* __restrict__ h,
                                 const float* __restrict__ lnw,
                                 int8_t* __restrict__ hq,
                                 float* __restrict__ sh) {
  __shared__ float reds[NTHREADS];
  __shared__ float redm[NTHREADS];
  __shared__ float s_rs, s_scale;
  const int row = blockIdx.x;
  const float* hr = h + (long)row * I_DIM;
  float ss = 0.f, m = 0.f;
  for (int i = threadIdx.x; i < I_DIM; i += NTHREADS) {
    float v = hr[i];
    ss += v * v;
    m = fmaxf(m, fabsf(lnw[i] * v));
  }
  reds[threadIdx.x] = ss;
  redm[threadIdx.x] = m;
  __syncthreads();
  for (int off = NTHREADS / 2; off > 0; off >>= 1) {
    if ((int)threadIdx.x < off) {
      reds[threadIdx.x] += reds[threadIdx.x + off];
      redm[threadIdx.x] = fmaxf(redm[threadIdx.x], redm[threadIdx.x + off]);
    }
    __syncthreads();
  }
  if (threadIdx.x == 0) {
    float var = reds[0] / (float)I_DIM;
    float rs = rsqrtf(var + 1e-6f);
    float amax = fmaxf(redm[0] * rs, 1e-5f);
    s_rs = rs;
    s_scale = 127.f / amax;
    sh[row] = amax / 127.f;
  }
  __syncthreads();
  const float rs = s_rs;
  const float sc = s_scale;
  int8_t* hqr = hq + (long)row * I_DIM;
  for (int i = threadIdx.x; i < I_DIM; i += NTHREADS) {
    float y = lnw[i] * hr[i] * rs;
    float q = rintf(y * sc);
    q = fminf(fmaxf(q, -128.f), 127.f);
    hqr[i] = (int8_t)(int)q;
  }
}

__global__ __launch_bounds__(NTHREADS)
void gemm_down_kernel(const int8_t* __restrict__ hq,
                      const int8_t* __restrict__ wqd,
                      const float* __restrict__ sh,
                      const float* __restrict__ wsc,
                      float* __restrict__ out) {
  __shared__ int8_t As[2][128 * 64];
  __shared__ int8_t Bs[2][64 * 64];

  const int tid = threadIdx.x;
  const int lane = tid & 31;
  const int wave = tid >> 5;
  const int wm = wave >> 1;
  const int wn = wave & 1;
  const long m0 = (long)blockIdx.y * 128;
  const long n0 = (long)blockIdx.x * 64;

  const int ra0 = (tid) >> 2,            qa0 = (tid) & 3;
  const int ra1 = (tid + NTHREADS) >> 2, qa1 = (tid + NTHREADS) & 3;
  const int rb  = tid >> 2,              qb  = tid & 3;

  v8i acc[2][2] = {};

  const int nsteps = I_DIM / 64;

  {
    const int k0 = 0;
    async_load_b128(&As[0][ra0 * 64 + qa0 * 16], &hq[(m0 + ra0) * (long)I_DIM + k0 + qa0 * 16]);
    async_load_b128(&As[0][ra1 * 64 + qa1 * 16], &hq[(m0 + ra1) * (long)I_DIM + k0 + qa1 * 16]);
    async_load_b128(&Bs[0][rb * 64 + qb * 16],  &wqd[(n0 + rb) * (long)I_DIM + k0 + qb * 16]);
  }

  for (int it = 0; it < nsteps; ++it) {
    const int cur = it & 1;
    if (it + 1 < nsteps) {
      const int nxt = cur ^ 1;
      const int k1 = (it + 1) * 64;
      async_load_b128(&As[nxt][ra0 * 64 + qa0 * 16], &hq[(m0 + ra0) * (long)I_DIM + k1 + qa0 * 16]);
      async_load_b128(&As[nxt][ra1 * 64 + qa1 * 16], &hq[(m0 + ra1) * (long)I_DIM + k1 + qa1 * 16]);
      async_load_b128(&Bs[nxt][rb * 64 + qb * 16],  &wqd[(n0 + rb) * (long)I_DIM + k1 + qb * 16]);
      ASYNC_WAIT(3);
    } else {
      ASYNC_WAIT(0);
    }
    __syncthreads();

    v8i a[2], b[2];
    #pragma unroll
    for (int mt = 0; mt < 2; ++mt) a[mt] = load_a_frag(As[cur], wm * 32 + mt * 16, lane);
    #pragma unroll
    for (int nt = 0; nt < 2; ++nt) b[nt] = load_b_frag(Bs[cur], wn * 32 + nt * 16, lane);
    #pragma unroll
    for (int mt = 0; mt < 2; ++mt)
      #pragma unroll
      for (int nt = 0; nt < 2; ++nt)
        acc[mt][nt] = __builtin_amdgcn_wmma_i32_16x16x64_iu8(
            true, a[mt], true, b[nt], acc[mt][nt], false, false);
    __syncthreads();
  }

  const float dqd = wsc[2];
  #pragma unroll
  for (int mt = 0; mt < 2; ++mt)
    #pragma unroll
    for (int nt = 0; nt < 2; ++nt) {
      const long rowbase = m0 + wm * 32 + mt * 16 + ((lane >> 4) << 3);
      const long col = n0 + wn * 32 + nt * 16 + (lane & 15);
      #pragma unroll
      for (int r = 0; r < 8; ++r) {
        long row = rowbase + r;
        out[row * (long)H_DIM + col] = (float)acc[mt][nt][r] * sh[row] * dqd;
      }
    }
}

extern "C" void kernel_launch(void* const* d_in, const int* in_sizes, int n_in,
                              void* d_out, int out_size, void* d_ws, size_t ws_size,
                              hipStream_t stream) {
  (void)in_sizes; (void)n_in; (void)out_size; (void)ws_size;
  const float* x      = (const float*)d_in[0];
  const float* w_gate = (const float*)d_in[1];
  const float* w_up   = (const float*)d_in[2];
  const float* w_down = (const float*)d_in[3];
  const float* lnw    = (const float*)d_in[4];
  float* out = (float*)d_out;

  const long NW = (long)I_DIM * H_DIM;
  char* ws = (char*)d_ws;
  size_t off = 0;
  int8_t* xq  = (int8_t*)(ws + off); off += (size_t)T_DIM * H_DIM;
  int8_t* wqg = (int8_t*)(ws + off); off += (size_t)NW;
  int8_t* wqu = (int8_t*)(ws + off); off += (size_t)NW;
  int8_t* wqd = (int8_t*)(ws + off); off += (size_t)NW;
  int8_t* hqb = (int8_t*)(ws + off); off += (size_t)T_DIM * I_DIM;
  float*  h   = (float*)(ws + off);  off += (size_t)T_DIM * I_DIM * 4;
  float*  sx  = (float*)(ws + off);  off += (size_t)T_DIM * 4;
  float*  sh  = (float*)(ws + off);  off += (size_t)T_DIM * 4;
  float*  wsc = (float*)(ws + off);  off += 64;
  float*  part= (float*)(ws + off);  off += (size_t)3 * 1024 * 4;

  absmean_partial_kernel<<<1024, NTHREADS, 0, stream>>>(w_gate, NW, part + 0);
  absmean_partial_kernel<<<1024, NTHREADS, 0, stream>>>(w_up,   NW, part + 1024);
  absmean_partial_kernel<<<1024, NTHREADS, 0, stream>>>(w_down, NW, part + 2048);
  absmean_final_kernel<<<3, NTHREADS, 0, stream>>>(part, NW, wsc);

  quant_w_kernel<<<4096, NTHREADS, 0, stream>>>(w_gate, wqg, NW, wsc + 0);
  quant_w_kernel<<<4096, NTHREADS, 0, stream>>>(w_up,   wqu, NW, wsc + 1);
  quant_w_kernel<<<4096, NTHREADS, 0, stream>>>(w_down, wqd, NW, wsc + 2);

  actquant_x_kernel<<<T_DIM, NTHREADS, 0, stream>>>(x, xq, sx);

  gemm_gateup_kernel<<<dim3(I_DIM / 64, T_DIM / 128), NTHREADS, 0, stream>>>(
      xq, wqg, wqu, sx, wsc, h);

  rms_quant_kernel<<<T_DIM, NTHREADS, 0, stream>>>(h, lnw, hqb, sh);

  gemm_down_kernel<<<dim3(H_DIM / 64, T_DIM / 128), NTHREADS, 0, stream>>>(
      hqb, wqd, sh, wsc, out);
}
